// AttEncode_4114578669881
// MI455X (gfx1250) — matrix operand order repaired and checked by reading, not compiled, not run on hardware
//
#include <hip/hip_runtime.h>

// ---------- types ----------
typedef __attribute__((ext_vector_type(16))) __bf16 v16bf;
typedef __attribute__((ext_vector_type(8)))  __bf16 v8bf;
typedef __attribute__((ext_vector_type(8)))  float  v8f;

__device__ __forceinline__ unsigned short f2bf(float f) {
    unsigned int u = __builtin_bit_cast(unsigned int, f);
    unsigned int r = u + 0x7FFFu + ((u >> 16) & 1u);   // round-to-nearest-even
    return (unsigned short)(r >> 16);
}

__device__ __forceinline__ v8f wmma_bf16(v16bf a, v16bf b, v8f c) {
    return __builtin_amdgcn_wmma_f32_16x16x32_bf16(
        /*neg_a=*/false, a, /*neg_b=*/false, b,
        /*c_mod=*/(short)0, c, /*reuse_a=*/false, /*reuse_b=*/false);
}

// ---------- problem constants ----------
#define Bn   64
#define Tn   2048
#define Dn   1024
#define Cn   1024
#define Fn   256
#define TT   28          // output time steps per tile
#define GR   32          // computed G rows (t0-2 .. t0+29)
#define GRP  36          // padded G rows (conv reads up to row 35)
#define SSTR 1032        // bf16 row stride (pad 8 -> 4-bank skew per row)
#define NTILES_PER_B 74  // ceil(2048/28)

// ================================================================
// Kernel 1: l2-normalize labels, write bf16 packed [D/32][C][32]
// ================================================================
__global__ __launch_bounds__(256) void k_labels(const float* __restrict__ lab,
                                                unsigned short* __restrict__ LNp) {
    const int c   = blockIdx.x;
    const int tid = threadIdx.x;
    const float* row = lab + (size_t)c * Dn;
    float4 v = *(const float4*)(row + tid * 4);
    float s = v.x * v.x + v.y * v.y + v.z * v.z + v.w * v.w;
    __shared__ float red[256];
    red[tid] = s;
    __syncthreads();
    for (int off = 128; off > 0; off >>= 1) {
        if (tid < off) red[tid] += red[tid + off];
        __syncthreads();
    }
    float x   = fmaxf(red[0], 1e-12f);
    float inv = rsqrtf(x);
    inv = inv * (1.5f - 0.5f * x * inv * inv);   // one NR step for fp32-grade rsqrt
    const int d0 = tid * 4;
    size_t base = (size_t)(d0 >> 5) * (Cn * 32) + (size_t)c * 32 + (d0 & 31);
    LNp[base + 0] = f2bf(v.x * inv);
    LNp[base + 1] = f2bf(v.y * inv);
    LNp[base + 2] = f2bf(v.z * inv);
    LNp[base + 3] = f2bf(v.w * inv);
}

// ================================================================
// Kernel 2: pack conv_w [K,C,F] -> bf16 [KC/32][F][32]
// ================================================================
__global__ __launch_bounds__(256) void k_packw(const float* __restrict__ cw,
                                               unsigned short* __restrict__ CWp) {
    const int o   = blockIdx.x * 256 + threadIdx.x;   // < 5120*256
    const int kk  = o & 31;
    const int f   = (o >> 5) & 255;
    const int blk = o >> 13;
    const int kc  = blk * 32 + kk;                     // flat (k*C + c)
    CWp[o] = f2bf(cw[(size_t)kc * Fn + f]);
}

// ================================================================
// Kernel 3: fused  G-GEMM -> LDS -> conv-GEMM -> relu/max -> att_v
//   grid = 64 * 74 tiles, 256 threads (8 waves)
// ================================================================
__global__ __launch_bounds__(256) void k_main(const float* __restrict__ sent,
                                              const unsigned short* __restrict__ LNp,
                                              const unsigned short* __restrict__ CWp,
                                              const float* __restrict__ convb,
                                              float* __restrict__ attv) {
    extern __shared__ unsigned short smem[];
    unsigned short* St   = smem;                       // 32 * 1032 bf16
    unsigned short* Gb   = smem + GR * SSTR;           // 36 * 1032 bf16
    float*          attw = (float*)(smem + (GR + GRP) * SSTR);  // 8*32 floats

    const int tid  = threadIdx.x;
    const int wave = tid >> 5;
    const int lane = tid & 31;
    const int m    = lane & 15;
    const int half = lane >> 4;

    const int b    = blockIdx.x / NTILES_PER_B;
    const int tile = blockIdx.x - b * NTILES_PER_B;
    const int t0   = tile * TT;

    // ---- stage sentence rows (t0-2 .. t0+29) into LDS as bf16 ----
    {
        const int col = tid * 4;                       // 0..1020
        for (int r = 0; r < GR; ++r) {
            int t = t0 - 2 + r;
            float4 v = make_float4(0.f, 0.f, 0.f, 0.f);
            if (t >= 0 && t < Tn)
                v = *(const float4*)(sent + ((size_t)b * Tn + t) * Dn + col);
            unsigned short* p = St + r * SSTR + col;
            p[0] = f2bf(v.x); p[1] = f2bf(v.y); p[2] = f2bf(v.z); p[3] = f2bf(v.w);
        }
        // zero halo rows 32..35 of G (read by discarded conv rows)
        for (int i = tid; i < 4 * SSTR; i += 256) {
            int r = i / SSTR, c = i - r * SSTR;
            Gb[(GR + r) * SSTR + c] = 0;
        }
    }
    __syncthreads();

    // ---- GEMM phase: G(32x1024) = St(32x1024) x LN^T(1024x1024) ----
    // wave owns columns [wave*128, wave*128+128): 2 M-tiles x 8 N-tiles
    const int nbase = wave << 7;
    v8f acc[2][8];
#pragma unroll
    for (int i = 0; i < 2; ++i)
#pragma unroll
        for (int j = 0; j < 8; ++j) acc[i][j] = (v8f){0.f,0.f,0.f,0.f,0.f,0.f,0.f,0.f};

    for (int d0 = 0; d0 < Dn; d0 += 32) {
        v16bf a[2];
#pragma unroll
        for (int mt = 0; mt < 2; ++mt) {
            const unsigned short* p = St + (mt * 16 + m) * SSTR + d0 + half * 16;
            v8bf lo = *(const v8bf*)p;
            v8bf hi = *(const v8bf*)(p + 8);
            a[mt] = __builtin_shufflevector(lo, hi, 0,1,2,3,4,5,6,7,8,9,10,11,12,13,14,15);
        }
        const unsigned short* q =
            LNp + (size_t)(d0 >> 5) * (Cn * 32) + (size_t)(nbase + m) * 32 + half * 16;
#pragma unroll
        for (int j = 0; j < 8; ++j) {
            v16bf bb = *(const v16bf*)(q + (size_t)j * (16 * 32));
            acc[0][j] = wmma_bf16(a[0], bb, acc[0][j]);
            acc[1][j] = wmma_bf16(a[1], bb, acc[1][j]);
        }
    }

    // ---- store G tile to LDS as bf16 (C-layout -> row major) ----
#pragma unroll
    for (int mt = 0; mt < 2; ++mt)
#pragma unroll
        for (int j = 0; j < 8; ++j)
#pragma unroll
            for (int jj = 0; jj < 8; ++jj)
                Gb[(mt * 16 + half * 8 + jj) * SSTR + nbase + j * 16 + m] =
                    f2bf(acc[mt][j][jj]);
    __syncthreads();

    // ---- conv phase: out(32x256) = Gshift(32x5120) x Wp(5120x256) ----
    // wave owns F columns [wave*32, wave*32+32): 2 M-tiles x 2 N-tiles
    v8f cacc[2][2];
#pragma unroll
    for (int i = 0; i < 2; ++i)
#pragma unroll
        for (int j = 0; j < 2; ++j) cacc[i][j] = (v8f){0.f,0.f,0.f,0.f,0.f,0.f,0.f,0.f};

    for (int kc0 = 0; kc0 < 5 * Cn; kc0 += 32) {
        const int k  = kc0 >> 10;        // conv tap (C==1024 -> 32 | 1024)
        const int c0 = kc0 & 1023;
        v16bf a[2];
#pragma unroll
        for (int mt = 0; mt < 2; ++mt) {
            const unsigned short* p = Gb + (mt * 16 + m + k) * SSTR + c0 + half * 16;
            v8bf lo = *(const v8bf*)p;
            v8bf hi = *(const v8bf*)(p + 8);
            a[mt] = __builtin_shufflevector(lo, hi, 0,1,2,3,4,5,6,7,8,9,10,11,12,13,14,15);
        }
        const unsigned short* q =
            CWp + (size_t)(kc0 >> 5) * (Fn * 32) + (size_t)(wave * 32 + m) * 32 + half * 16;
#pragma unroll
        for (int j = 0; j < 2; ++j) {
            v16bf bb = *(const v16bf*)(q + (size_t)j * (16 * 32));
            cacc[0][j] = wmma_bf16(a[0], bb, cacc[0][j]);
            cacc[1][j] = wmma_bf16(a[1], bb, cacc[1][j]);
        }
    }

    // ---- bias + relu + max over F ----
    const float bias0 = convb[wave * 32 + m];
    const float bias1 = convb[wave * 32 + 16 + m];
#pragma unroll
    for (int mt = 0; mt < 2; ++mt) {
#pragma unroll
        for (int jj = 0; jj < 8; ++jj) {
            float v = fmaxf(fmaxf(cacc[mt][0][jj] + bias0, 0.f),
                            fmaxf(cacc[mt][1][jj] + bias1, 0.f));
            // reduce max across the 16 lanes holding this row (stays in half)
            v = fmaxf(v, __shfl_xor(v, 1, 32));
            v = fmaxf(v, __shfl_xor(v, 2, 32));
            v = fmaxf(v, __shfl_xor(v, 4, 32));
            v = fmaxf(v, __shfl_xor(v, 8, 32));
            if (m == 0) attw[wave * 32 + mt * 16 + half * 8 + jj] = v;
        }
    }
    __syncthreads();

    if (tid < 32) {
        float mx = attw[tid];
#pragma unroll
        for (int w = 1; w < 8; ++w) mx = fmaxf(mx, attw[w * 32 + tid]);
        const int t = t0 + tid;
        if (tid < TT && t < Tn) attv[(size_t)b * Tn + t] = mx;
    }
}

// ================================================================
// Kernel 4: H_enc[b,d] = sum_t sentence[b,t,d] * att_v[b,t]  (fp32, deterministic)
// ================================================================
__global__ __launch_bounds__(256) void k_henc(const float* __restrict__ sent,
                                              const float* __restrict__ attv,
                                              float* __restrict__ out) {
    const int b = blockIdx.x >> 2;
    const int d = ((blockIdx.x & 3) << 8) + threadIdx.x;
    const float* sb = sent + (size_t)b * Tn * Dn + d;
    const float* ab = attv + (size_t)b * Tn;
    float acc = 0.f;
#pragma unroll 4
    for (int t = 0; t < Tn; ++t)
        acc = fmaf(ab[t], sb[(size_t)t * Dn], acc);
    out[(size_t)b * Dn + d] = acc;
}

// ================================================================
extern "C" void kernel_launch(void* const* d_in, const int* in_sizes, int n_in,
                              void* d_out, int out_size, void* d_ws, size_t ws_size,
                              hipStream_t stream) {
    const float* sent = (const float*)d_in[0];   // [64,2048,1024]
    const float* lab  = (const float*)d_in[1];   // [1024,1024]
    const float* cw   = (const float*)d_in[2];   // [5,1024,256]
    const float* cb   = (const float*)d_in[3];   // [256]
    float* out = (float*)d_out;                  // [64,1024]

    char* ws = (char*)d_ws;
    unsigned short* LNp  = (unsigned short*)ws;                        // 2,097,152 B
    unsigned short* CWp  = (unsigned short*)(ws + 2097152);            // 2,621,440 B
    float*          attv = (float*)(ws + 2097152 + 2621440);           //   524,288 B

    k_labels<<<Cn, 256, 0, stream>>>(lab, LNp);
    k_packw<<<(5 * Cn * Fn) / 256, 256, 0, stream>>>(cw, CWp);

    const size_t smem = (size_t)(GR + GRP) * SSTR * sizeof(unsigned short)
                      + 8 * 32 * sizeof(float);     // 141,376 B
    k_main<<<Bn * NTILES_PER_B, 256, smem, stream>>>(sent, LNp, CWp, cb, attv);

    k_henc<<<Bn * (Dn / 256), 256, 0, stream>>>(sent, attv, out);
}
